// VisualBert_REX_73186242724172
// MI455X (gfx1250) — matrix operand order repaired and checked by reading, not compiled, not run on hardware
//
#include <hip/hip_runtime.h>
#include <hip/hip_bf16.h>

#define Bn   256
#define ROIn 36
#define IMGn 2048
#define HIDn 768
#define SEQn 56
#define STEPSn 12
#define Vn   2001
#define ANSn 2000

typedef __attribute__((ext_vector_type(16))) __bf16 v16bf;
typedef __attribute__((ext_vector_type(8)))  __bf16 v8bf;
typedef __attribute__((ext_vector_type(8)))  float  v8f;

__device__ __forceinline__ v8f zero8() {
    v8f z;
    #pragma unroll
    for (int i = 0; i < 8; ++i) z[i] = 0.0f;
    return z;
}

// A-fragment loader: contiguous k order (element s -> k = off + s), same as the B loads.
// bf16 A: two contiguous 16B vector loads.  f32 A: scalar load + convert.
template <typename TA>
__device__ __forceinline__ v16bf load_a(const TA* __restrict__ Arow, int off) {
    v16bf a;
    if constexpr (sizeof(TA) == 2) {
        v8bf lo = *(const v8bf*)(Arow + off);
        v8bf hi = *(const v8bf*)(Arow + off + 8);
        #pragma unroll
        for (int e = 0; e < 8; ++e) { a[e] = lo[e]; a[e + 8] = hi[e]; }
    } else {
        #pragma unroll
        for (int e = 0; e < 8; ++e) {
            a[e]     = (__bf16)Arow[off + e];
            a[e + 8] = (__bf16)Arow[off + 8 + e];
        }
    }
    return a;
}

__device__ __forceinline__ void store_tile(v8f acc, int mbase, int n, float* C, int ldc,
                                           int N, const float* bias, int accum, int epi,
                                           int half) {
    if (n >= N) return;                       // only ragged-N guard; M always exact
    float bb = bias ? bias[n] : 0.0f;
    #pragma unroll
    for (int i = 0; i < 8; ++i) {
        int m = mbase + i + half * 8;          // C/D layout: VGPR i -> M=i (+8 hi half)
        size_t idx = (size_t)m * ldc + n;
        float v = acc[i] + bb;
        if (accum) v += C[idx];
        if (epi == 1)      v = tanhf(v);
        else if (epi == 2) v = 1.0f / (1.0f + __expf(-v));
        C[idx] = v;
    }
}

// ---------------------------------------------------------------------------
// WMMA GEMM, 2x2 register blocking: C[M,N] (+)= A[M,K] * Bt[N,K](bf16) + bias
// block = 128 thr = 4 waves; each wave -> 32x32 patch (4 wmma / k-step).
// grid = (ceil(N/64), M/64).  M % 64 == 0 required; N ragged allowed.
// Ragged-N is handled branch-free: OOB column loads are clamped to row N-1
// (always a valid address); their results are simply never stored.
// ---------------------------------------------------------------------------
template <typename TA>
__global__ void __launch_bounds__(128)
gemm_wmma(const TA* __restrict__ A, int lda,
          const __bf16* __restrict__ Bt,
          float* __restrict__ C, int ldc,
          int M, int N, int K,
          const float* __restrict__ bias, int accum, int epi) {
    const int wave = threadIdx.x >> 5;
    const int lane = threadIdx.x & 31;
    const int l15  = lane & 15;
    const int half = lane >> 4;
    const int m0 = blockIdx.y * 64 + (wave >> 1) * 32;
    const int n0 = blockIdx.x * 64 + (wave & 1) * 32;

    const int akb = half * 16;
    const int bkb = half * 16;
    const TA* Arow0 = A + (size_t)(m0 + l15) * lda;
    const TA* Arow1 = A + (size_t)(m0 + 16 + l15) * lda;
    const int bn0 = n0 + l15, bn1 = n0 + 16 + l15;
    const __bf16* Brow0 = Bt + (size_t)(bn0 < N ? bn0 : N - 1) * K;  // clamp, no branch
    const __bf16* Brow1 = Bt + (size_t)(bn1 < N ? bn1 : N - 1) * K;

    v8f acc00 = zero8(), acc01 = zero8(), acc10 = zero8(), acc11 = zero8();

    for (int kk = 0; kk < K; kk += 32) {
        v16bf a0 = load_a<TA>(Arow0, kk + akb);
        v16bf a1 = load_a<TA>(Arow1, kk + akb);
        v16bf b0 = *(const v16bf*)(Brow0 + kk + bkb);
        v16bf b1 = *(const v16bf*)(Brow1 + kk + bkb);
        acc00 = __builtin_amdgcn_wmma_f32_16x16x32_bf16(false, a0, false, b0, (short)0, acc00, false, false);
        acc01 = __builtin_amdgcn_wmma_f32_16x16x32_bf16(false, a0, false, b1, (short)0, acc01, false, false);
        acc10 = __builtin_amdgcn_wmma_f32_16x16x32_bf16(false, a1, false, b0, (short)0, acc10, false, false);
        acc11 = __builtin_amdgcn_wmma_f32_16x16x32_bf16(false, a1, false, b1, (short)0, acc11, false, false);
    }

    store_tile(acc00, m0,      n0 + l15,      C, ldc, N, bias, accum, epi, half);
    store_tile(acc01, m0,      n0 + 16 + l15, C, ldc, N, bias, accum, epi, half);
    store_tile(acc10, m0 + 16, n0 + l15,      C, ldc, N, bias, accum, epi, half);
    store_tile(acc11, m0 + 16, n0 + 16 + l15, C, ldc, N, bias, accum, epi, half);
}

// ---------------------------------------------------------------------------
// Weight pack: W (K x N f32) -> Wt (N x K bf16)
// ---------------------------------------------------------------------------
__global__ void pack_wT(const float* __restrict__ W, __bf16* __restrict__ Wt,
                        int K, int N) {
    int idx = blockIdx.x * blockDim.x + threadIdx.x;
    if (idx >= K * N) return;
    int k = idx / N, n = idx % N;
    Wt[(size_t)n * K + k] = (__bf16)W[idx];
}

// ---------------------------------------------------------------------------
// Pointwise / reductions
// ---------------------------------------------------------------------------
__global__ void fill_zero(float* p, int n) {
    int i = blockIdx.x * blockDim.x + threadIdx.x;
    if (i < n) p[i] = 0.0f;
}
__global__ void fill_zero_b(__bf16* p, int n) {
    int i = blockIdx.x * blockDim.x + threadIdx.x;
    if (i < n) p[i] = (__bf16)0.0f;
}

__global__ void fuse_mul(float* fuse, const float* __restrict__ qatt) {
    int i = blockIdx.x * blockDim.x + threadIdx.x;
    if (i < Bn * ROIn * HIDn) {
        int b = i / (ROIn * HIDn);
        int d = i % HIDn;
        fuse[i] *= qatt[b * HIDn + d];
    }
}

__global__ void fuse_reduce(const float* __restrict__ fuse,
                            float* __restrict__ mean, float* __restrict__ sum) {
    int i = blockIdx.x * blockDim.x + threadIdx.x;
    if (i < Bn * HIDn) {
        int b = i / HIDn, d = i % HIDn;
        const float* f = fuse + (size_t)b * ROIn * HIDn + d;
        float s = 0.0f;
        for (int r = 0; r < ROIn; ++r) s += f[(size_t)r * HIDn];
        sum[i]  = s;
        mean[i] = s * (1.0f / (float)ROIn);
    }
}

__global__ void assemble_x1b(__bf16* __restrict__ x1, const float* __restrict__ f,
                             const float* __restrict__ h, const float* __restrict__ p) {
    int i = blockIdx.x * blockDim.x + threadIdx.x;
    if (i >= Bn * 3 * HIDn) return;
    int b = i / (3 * HIDn), c = i % (3 * HIDn);
    float v;
    if (c < HIDn)          v = f[b * HIDn + c];
    else if (c < 2 * HIDn) v = h[b * HIDn + c - HIDn];
    else                   v = p[b * HIDn + c - 2 * HIDn];
    x1[i] = (__bf16)v;
}

__global__ void assemble_x2b(__bf16* __restrict__ x2, const float* __restrict__ venc,
                             const float* __restrict__ qenc, const float* __restrict__ h1) {
    int i = blockIdx.x * blockDim.x + threadIdx.x;
    if (i >= Bn * 2 * HIDn) return;
    int b = i / (2 * HIDn), c = i % (2 * HIDn);
    float v = (c < HIDn) ? venc[b * HIDn + c] * qenc[b * HIDn + c]
                         : h1[b * HIDn + c - HIDn];
    x2[i] = (__bf16)v;
}

// z = sig(gx[:,0:H]+gh[:,0:H]);  r = sig(gx[:,H:2H]+gh[:,H:2H]);  rhb = bf16(r*h)
__global__ void gru_zr(const float* __restrict__ gx, const float* __restrict__ gh,
                       const float* __restrict__ h,
                       float* __restrict__ z, __bf16* __restrict__ rhb) {
    int i = blockIdx.x * blockDim.x + threadIdx.x;
    if (i >= Bn * HIDn) return;
    int b = i / HIDn, d = i % HIDn;
    float zz = 1.0f / (1.0f + __expf(-(gx[(size_t)b * 3 * HIDn + d] + gh[(size_t)b * 2 * HIDn + d])));
    float rr = 1.0f / (1.0f + __expf(-(gx[(size_t)b * 3 * HIDn + HIDn + d] + gh[(size_t)b * 2 * HIDn + HIDn + d])));
    z[i]   = zz;
    rhb[i] = (__bf16)(rr * h[i]);
}

// h' = (1-z)*h + z*m ; m = gx[:,2H:3H] (tanh already applied by GEMM epilogue)
__global__ void gru_out2(float* __restrict__ ho, __bf16* __restrict__ hob,
                         const float* __restrict__ z, const float* __restrict__ gx,
                         const float* __restrict__ h) {
    int i = blockIdx.x * blockDim.x + threadIdx.x;
    if (i >= Bn * HIDn) return;
    int b = i / HIDn, d = i % HIDn;
    float m = gx[(size_t)b * 3 * HIDn + 2 * HIDn + d];
    float zz = z[i];
    float v = (1.0f - zz) * h[i] + zz * m;
    ho[i]  = v;
    hob[i] = (__bf16)v;
}

// scores[b,r] = sum_d tanh(hq[b,d] + fuse[b,r,d]) * w_att[d] + b_att_s
__global__ void att_score(const float* __restrict__ hq, const float* __restrict__ fuse,
                          const float* __restrict__ w_att, const float* __restrict__ b_s,
                          float* __restrict__ scores) {
    int b = blockIdx.y, r = blockIdx.x, tid = threadIdx.x;
    const float* f = fuse + ((size_t)b * ROIn + r) * HIDn;
    const float* q = hq + (size_t)b * HIDn;
    float s = 0.0f;
    for (int d = tid; d < HIDn; d += blockDim.x) s += tanhf(q[d] + f[d]) * w_att[d];
    __shared__ float red[256];
    red[tid] = s; __syncthreads();
    for (int o = blockDim.x >> 1; o > 0; o >>= 1) {
        if (tid < o) red[tid] += red[tid + o];
        __syncthreads();
    }
    if (tid == 0) scores[b * ROIn + r] = red[0] + b_s[0];
}

// softmax over ROI, then attx[b,:] = bf16( sum_r w[r]*img[b,r,:] )
__global__ void softmax_attx(const float* __restrict__ scores,
                             const float* __restrict__ img, __bf16* __restrict__ attx) {
    int b = blockIdx.x, tid = threadIdx.x;
    __shared__ float w[ROIn];
    if (tid == 0) {
        float mx = -3.4e38f;
        for (int r = 0; r < ROIn; ++r) mx = fmaxf(mx, scores[b * ROIn + r]);
        float s = 0.0f;
        for (int r = 0; r < ROIn; ++r) { w[r] = __expf(scores[b * ROIn + r] - mx); s += w[r]; }
        float inv = 1.0f / s;
        for (int r = 0; r < ROIn; ++r) w[r] *= inv;
    }
    __syncthreads();
    for (int d = tid; d < IMGn; d += blockDim.x) {
        float acc = 0.0f;
        for (int r = 0; r < ROIn; ++r)
            acc += w[r] * img[((size_t)b * ROIn + r) * IMGn + d];
        attx[(size_t)b * IMGn + d] = (__bf16)acc;
    }
}

__global__ void sim_score(const float* __restrict__ h2, const float* __restrict__ bert,
                          float* __restrict__ out) {
    int b = blockIdx.y, r = blockIdx.x, tid = threadIdx.x;
    const float* vf = bert + ((size_t)b * SEQn + (SEQn - ROIn) + r) * HIDn;
    const float* h  = h2 + (size_t)b * HIDn;
    float s = 0.0f;
    for (int d = tid; d < HIDn; d += blockDim.x) s += h[d] * vf[d];
    __shared__ float red[256];
    red[tid] = s; __syncthreads();
    for (int o = blockDim.x >> 1; o > 0; o >>= 1) {
        if (tid < o) red[tid] += red[tid + o];
        __syncthreads();
    }
    if (tid == 0) out[b * ROIn + r] = red[0];
}

__global__ void row_softmax(float* __restrict__ x, int n) {
    int b = blockIdx.x, tid = threadIdx.x, bs = blockDim.x;
    float* row = x + (size_t)b * n;
    __shared__ float red[256];
    float mx = -3.4e38f;
    for (int i = tid; i < n; i += bs) mx = fmaxf(mx, row[i]);
    red[tid] = mx; __syncthreads();
    for (int o = bs >> 1; o > 0; o >>= 1) {
        if (tid < o) red[tid] = fmaxf(red[tid], red[tid + o]);
        __syncthreads();
    }
    mx = red[0]; __syncthreads();
    float s = 0.0f;
    for (int i = tid; i < n; i += bs) { float e = __expf(row[i] - mx); row[i] = e; s += e; }
    red[tid] = s; __syncthreads();
    for (int o = bs >> 1; o > 0; o >>= 1) {
        if (tid < o) red[tid] += red[tid + o];
        __syncthreads();
    }
    float inv = 1.0f / red[0];
    for (int i = tid; i < n; i += bs) row[i] *= inv;
}

__global__ void gate_dot(const float* __restrict__ h2, const float* __restrict__ wg,
                         const float* __restrict__ bg, float* __restrict__ gate) {
    int b = blockIdx.x, tid = threadIdx.x;
    float s = 0.0f;
    for (int d = tid; d < HIDn; d += blockDim.x) s += h2[(size_t)b * HIDn + d] * wg[d];
    __shared__ float red[256];
    red[tid] = s; __syncthreads();
    for (int o = blockDim.x >> 1; o > 0; o >>= 1) {
        if (tid < o) red[tid] += red[tid + o];
        __syncthreads();
    }
    if (tid == 0) gate[b] = 1.0f / (1.0f + __expf(-(red[0] + bg[0])));
}

__global__ void combine_pred(const float* __restrict__ pred, const float* __restrict__ sim,
                             const float* __restrict__ SM, const float* __restrict__ gate,
                             float* __restrict__ out_sent, int t) {
    int b = blockIdx.y;
    int v = blockIdx.x * blockDim.x + threadIdx.x;
    __shared__ float sw[ROIn];
    if (threadIdx.x < ROIn) sw[threadIdx.x] = sim[b * ROIn + threadIdx.x];
    __syncthreads();
    if (v < Vn) {
        const float* smrow = SM + (size_t)v * ROIn;
        float sp = 0.0f;
        for (int r = 0; r < ROIn; ++r) sp += sw[r] * smrow[r];
        float g = gate[b];
        out_sent[((size_t)b * STEPSn + t) * Vn + v] =
            g * pred[(size_t)b * Vn + v] + (1.0f - g) * sp;
    }
}

__global__ void argmax_embed(const float* __restrict__ out_sent, int t,
                             const float* __restrict__ emb, float* __restrict__ prev) {
    int b = blockIdx.x, tid = threadIdx.x;
    const float* row = out_sent + ((size_t)b * STEPSn + t) * Vn;
    float best = -3.4e38f; int bi = Vn;
    for (int v = tid; v < Vn; v += blockDim.x) {
        float x = row[v];
        if (x > best) { best = x; bi = v; }
    }
    __shared__ float bv[256]; __shared__ int bidx[256];
    bv[tid] = best; bidx[tid] = bi; __syncthreads();
    for (int o = blockDim.x >> 1; o > 0; o >>= 1) {
        if (tid < o) {
            if (bv[tid + o] > bv[tid] ||
                (bv[tid + o] == bv[tid] && bidx[tid + o] < bidx[tid])) {
                bv[tid] = bv[tid + o]; bidx[tid] = bidx[tid + o];
            }
        }
        __syncthreads();
    }
    __shared__ int widx;
    if (tid == 0) widx = bidx[0];
    __syncthreads();
    const float* e = emb + (size_t)widx * HIDn;
    for (int d = tid; d < HIDn; d += blockDim.x) prev[(size_t)b * HIDn + d] = e[d];
}

// ---------------------------------------------------------------------------
extern "C" void kernel_launch(void* const* d_in, const int* in_sizes, int n_in,
                              void* d_out, int out_size, void* d_ws, size_t ws_size,
                              hipStream_t stream) {
    (void)in_sizes; (void)n_in; (void)out_size; (void)ws_size;

    const float* img     = (const float*)d_in[0];
    const float* bert    = (const float*)d_in[1];
    const float* W_att_v = (const float*)d_in[2];
    const float* b_att_v = (const float*)d_in[3];
    const float* W_att_q = (const float*)d_in[4];
    const float* b_att_q = (const float*)d_in[5];
    const float* W_att_h = (const float*)d_in[6];
    const float* b_att_h = (const float*)d_in[7];
    const float* w_att   = (const float*)d_in[8];
    const float* b_att_s = (const float*)d_in[9];
    const float* Wx_att  = (const float*)d_in[10];
    const float* bx_att  = (const float*)d_in[11];
    const float* Wh_att  = (const float*)d_in[12];
    const float* bh_att  = (const float*)d_in[13];
    const float* W_qfc   = (const float*)d_in[14];
    const float* b_qfc   = (const float*)d_in[15];
    const float* W_vfc   = (const float*)d_in[16];
    const float* b_vfc   = (const float*)d_in[17];
    const float* Wx_lang = (const float*)d_in[18];
    const float* bx_lang = (const float*)d_in[19];
    const float* Wh_lang = (const float*)d_in[20];
    const float* bh_lang = (const float*)d_in[21];
    const float* W_lf    = (const float*)d_in[22];
    const float* b_lf    = (const float*)d_in[23];
    const float* W_gate  = (const float*)d_in[24];
    const float* b_gate  = (const float*)d_in[25];
    const float* W_ans   = (const float*)d_in[26];
    const float* b_ans   = (const float*)d_in[27];
    const float* exp_emb = (const float*)d_in[28];
    const float* SM      = (const float*)d_in[29];

    float* out_ans  = (float*)d_out;
    float* out_sent = out_ans + (size_t)Bn * ANSn;

    // ---- workspace: bf16 region (weights + activations), then f32 region ----
    __bf16* bfb = (__bf16*)d_ws;
    size_t bc = 0;
    auto balloc = [&](size_t n) { __bf16* p = bfb + bc; bc += (n + 15) & ~(size_t)15; return p; };
    __bf16* pWv   = balloc((size_t)IMGn * HIDn);
    __bf16* pWq   = balloc((size_t)HIDn * HIDn);
    __bf16* pWah  = balloc((size_t)HIDn * HIDn);
    __bf16* pWxa  = balloc((size_t)3 * 3 * HIDn * HIDn);   // 2304 x 2304 (3 gates stacked, N-major)
    __bf16* pWha  = balloc((size_t)3 * HIDn * HIDn);       // 2304 x 768
    __bf16* pWqfc = balloc((size_t)HIDn * HIDn);
    __bf16* pWvfc = balloc((size_t)IMGn * HIDn);
    __bf16* pWxl  = balloc((size_t)3 * 2 * HIDn * HIDn);   // 2304 x 1536
    __bf16* pWhl  = balloc((size_t)3 * HIDn * HIDn);
    __bf16* pWlf  = balloc((size_t)HIDn * Vn);
    __bf16* pWans = balloc((size_t)HIDn * ANSn);
    __bf16* x1b   = balloc((size_t)Bn * 3 * HIDn);
    __bf16* x2b   = balloc((size_t)Bn * 2 * HIDn);
    __bf16* rhb   = balloc((size_t)Bn * HIDn);
    __bf16* h1b   = balloc((size_t)Bn * HIDn);
    __bf16* h2b   = balloc((size_t)Bn * HIDn);
    __bf16* attxb = balloc((size_t)Bn * IMGn);

    float* fbase = (float*)(bfb + bc);
    size_t fc = 0;
    auto falloc = [&](size_t n) { float* p = fbase + fc; fc += (n + 15) & ~(size_t)15; return p; };
    float* fuse  = falloc((size_t)Bn * ROIn * HIDn);
    float* qatt  = falloc((size_t)Bn * HIDn);
    float* qenc  = falloc((size_t)Bn * HIDn);
    float* fmean = falloc((size_t)Bn * HIDn);
    float* fsum  = falloc((size_t)Bn * HIDn);
    float* h1    = falloc((size_t)Bn * HIDn);
    float* h2    = falloc((size_t)Bn * HIDn);
    float* prev  = falloc((size_t)Bn * HIDn);
    float* gx    = falloc((size_t)Bn * 3 * HIDn);
    float* gh    = falloc((size_t)Bn * 2 * HIDn);
    float* zbuf  = falloc((size_t)Bn * HIDn);
    float* hq    = falloc((size_t)Bn * HIDn);
    float* scores= falloc((size_t)Bn * ROIn);
    float* venc  = falloc((size_t)Bn * HIDn);
    float* pred  = falloc((size_t)Bn * Vn);
    float* sim   = falloc((size_t)Bn * ROIn);
    float* gate  = falloc((size_t)Bn);

    auto pack = [&](const float* W, __bf16* dst, int K, int N) {
        int total = K * N;
        pack_wT<<<dim3((total + 255) / 256), 256, 0, stream>>>(W, dst, K, N);
    };
    auto gemmB = [&](const __bf16* A, int lda, const __bf16* Bt, float* C, int ldc,
                     int M, int N, int K, const float* bias, int accum, int epi) {
        dim3 g((N + 63) / 64, M / 64);
        gemm_wmma<__bf16><<<g, 128, 0, stream>>>(A, lda, Bt, C, ldc, M, N, K, bias, accum, epi);
    };
    auto gemmF = [&](const float* A, int lda, const __bf16* Bt, float* C, int ldc,
                     int M, int N, int K, const float* bias, int accum, int epi) {
        dim3 g((N + 63) / 64, M / 64);
        gemm_wmma<float><<<g, 128, 0, stream>>>(A, lda, Bt, C, ldc, M, N, K, bias, accum, epi);
    };
    const int NE = Bn * HIDn;
    auto pw = [&](int n) { return dim3((n + 255) / 256); };

    // ---- pack all weights (transposed bf16; GRU gates stacked N-major) ----
    pack(W_att_v, pWv,  IMGn, HIDn);
    pack(W_att_q, pWq,  HIDn, HIDn);
    pack(W_att_h, pWah, HIDn, HIDn);
    for (int i = 0; i < 3; ++i)
        pack(Wx_att + (size_t)i * 3 * HIDn * HIDn, pWxa + (size_t)i * 3 * HIDn * HIDn, 3 * HIDn, HIDn);
    for (int i = 0; i < 3; ++i)
        pack(Wh_att + (size_t)i * HIDn * HIDn, pWha + (size_t)i * HIDn * HIDn, HIDn, HIDn);
    pack(W_qfc, pWqfc, HIDn, HIDn);
    pack(W_vfc, pWvfc, IMGn, HIDn);
    for (int i = 0; i < 3; ++i)
        pack(Wx_lang + (size_t)i * 2 * HIDn * HIDn, pWxl + (size_t)i * 2 * HIDn * HIDn, 2 * HIDn, HIDn);
    for (int i = 0; i < 3; ++i)
        pack(Wh_lang + (size_t)i * HIDn * HIDn, pWhl + (size_t)i * HIDn * HIDn, HIDn, HIDn);
    pack(W_lf,  pWlf,  HIDn, Vn);
    pack(W_ans, pWans, HIDn, ANSn);

    // ---- state init ----
    fill_zero<<<pw(NE), 256, 0, stream>>>(h2, NE);
    fill_zero<<<pw(NE), 256, 0, stream>>>(prev, NE);
    fill_zero_b<<<pw(NE), 256, 0, stream>>>(h2b, NE);

    // ---- pre-loop (f32-A GEMM path; these run once) ----
    gemmF(img, IMGn, pWv, fuse, HIDn, Bn * ROIn, HIDn, IMGn, b_att_v, 0, 1);      // v_att
    gemmF(bert, SEQn * HIDn, pWq, qatt, HIDn, Bn, HIDn, HIDn, b_att_q, 0, 1);     // q_att (cls)
    fuse_mul<<<pw(Bn * ROIn * HIDn), 256, 0, stream>>>(fuse, qatt);
    fuse_reduce<<<pw(NE), 256, 0, stream>>>(fuse, fmean, fsum);
    gemmF(bert, SEQn * HIDn, pWqfc, qenc, HIDn, Bn, HIDn, HIDn, b_qfc, 0, 1);     // q_enc
    gemmF(bert, SEQn * HIDn, pWans, out_ans, ANSn, Bn, ANSn, HIDn, b_ans, 0, 0);  // output_ans

    // ---- recurrent steps ----
    for (int t = 0; t < STEPSn; ++t) {
        assemble_x1b<<<pw(Bn * 3 * HIDn), 256, 0, stream>>>(x1b, (t == 0) ? fmean : fsum, h2, prev);

        // GRU att -> h1 : fused-gate GEMMs (gate2 uses (r*h)@Wh[2])
        gemmB(x1b, 3 * HIDn, pWxa, gx, 3 * HIDn, Bn, 3 * HIDn, 3 * HIDn, bx_att, 0, 0);
        gemmB(h2b, HIDn,     pWha, gh, 2 * HIDn, Bn, 2 * HIDn, HIDn,     bh_att, 0, 0);
        gru_zr<<<pw(NE), 256, 0, stream>>>(gx, gh, h2, zbuf, rhb);
        gemmB(rhb, HIDn, pWha + (size_t)2 * HIDn * HIDn, gx + 2 * HIDn, 3 * HIDn,
              Bn, HIDn, HIDn, bh_att + 2 * HIDn, 1, 1);
        gru_out2<<<pw(NE), 256, 0, stream>>>(h1, h1b, zbuf, gx, h2);

        // visual attention
        gemmB(h1b, HIDn, pWah, hq, HIDn, Bn, HIDn, HIDn, b_att_h, 0, 0);
        att_score<<<dim3(ROIn, Bn), 256, 0, stream>>>(hq, fuse, w_att, b_att_s, scores);
        softmax_attx<<<Bn, 256, 0, stream>>>(scores, img, attxb);
        gemmB(attxb, IMGn, pWvfc, venc, HIDn, Bn, HIDn, IMGn, b_vfc, 0, 1);       // v_enc
        assemble_x2b<<<pw(Bn * 2 * HIDn), 256, 0, stream>>>(x2b, venc, qenc, h1);

        // GRU lang -> h2 (in place)
        gemmB(x2b, 2 * HIDn, pWxl, gx, 3 * HIDn, Bn, 3 * HIDn, 2 * HIDn, bx_lang, 0, 0);
        gemmB(h2b, HIDn,     pWhl, gh, 2 * HIDn, Bn, 2 * HIDn, HIDn,     bh_lang, 0, 0);
        gru_zr<<<pw(NE), 256, 0, stream>>>(gx, gh, h2, zbuf, rhb);
        gemmB(rhb, HIDn, pWhl + (size_t)2 * HIDn * HIDn, gx + 2 * HIDn, 3 * HIDn,
              Bn, HIDn, HIDn, bh_lang + 2 * HIDn, 1, 1);
        gru_out2<<<pw(NE), 256, 0, stream>>>(h2, h2b, zbuf, gx, h2);

        // prediction head
        gemmB(h2b, HIDn, pWlf, pred, Vn, Bn, Vn, HIDn, b_lf, 0, 0);
        row_softmax<<<Bn, 256, 0, stream>>>(pred, Vn);
        gate_dot<<<Bn, 256, 0, stream>>>(h2, W_gate, b_gate, gate);
        sim_score<<<dim3(ROIn, Bn), 256, 0, stream>>>(h2, bert, sim);
        row_softmax<<<Bn, 64, 0, stream>>>(sim, ROIn);
        combine_pred<<<dim3((Vn + 255) / 256, Bn), 256, 0, stream>>>(pred, sim, SM, gate, out_sent, t);
        argmax_embed<<<Bn, 256, 0, stream>>>(out_sent, t, exp_emb, prev);
    }
}
